// CausalMultiHeadSelfAttention_8031588843821
// MI455X (gfx1250) — matrix operand order repaired and checked by reading, not compiled, not run on hardware
//
#include <hip/hip_runtime.h>
#include <hip/hip_bf16.h>

#define B_  2
#define S_  2048
#define D_  2048
#define H_  16
#define DK_ 128
#define PAD 8   // bf16 elements = 16 bytes of row padding (bank spread)

typedef __bf16 bf16_t;
typedef __attribute__((ext_vector_type(16))) __bf16 v16bf;
typedef __attribute__((ext_vector_type(8)))  __bf16 v8bf;
typedef __attribute__((ext_vector_type(8)))  float  v8f;

static __device__ __forceinline__ v16bf pack16(v8bf lo, v8bf hi) {
  v16bf r;
#pragma unroll
  for (int i = 0; i < 8; ++i) { r[i] = lo[i]; r[i + 8] = hi[i]; }
  return r;
}
static __device__ __forceinline__ v16bf load16(const bf16_t* p) {
  v8bf lo = *(const v8bf*)p;
  v8bf hi = *(const v8bf*)(p + 8);
  return pack16(lo, hi);
}
static __device__ __forceinline__ v8f wmma_bf16(v16bf a, v16bf b, v8f c) {
  return __builtin_amdgcn_wmma_f32_16x16x32_bf16(false, a, false, b, (short)0, c,
                                                 false, false);
}

// --- CDNA5 async global->LDS copy (ASYNCcnt path), 16 bytes per lane --------
static __device__ __forceinline__ void async16(unsigned lds_off, const void* g) {
  asm volatile("global_load_async_to_lds_b128 %0, %1, off"
               :: "v"(lds_off), "v"((unsigned long long)(uintptr_t)g)
               : "memory");
}
static __device__ __forceinline__ void wait_async0() {
  asm volatile("s_wait_asynccnt 0" ::: "memory");
}

// ---------------------------------------------------------------------------
// f32 -> bf16 conversion
// ---------------------------------------------------------------------------
__global__ void __launch_bounds__(256)
cvt_f32_bf16(const float* __restrict__ src, bf16_t* __restrict__ dst, int n) {
  int i = blockIdx.x * 256 + threadIdx.x;
  if (i < n) dst[i] = (bf16_t)src[i];
}

// ---------------------------------------------------------------------------
// QKV projection: C = X (4096x2048) * W^T. 8 waves / block, 128x128 macro-tile,
// double-buffered async LDS staging of 128x64 A/B tiles (k-step 64 = two WMMA
// k-chunks per barrier). B-fragment loads are software-pipelined one ahead of
// the consuming WMMA so waits are counted, not dscnt==0.
// Epilogue: Q/K fuse RoPE -> [B*H][S][DK]; V stored transposed [B*H][DK][S].
// ---------------------------------------------------------------------------
__global__ void __launch_bounds__(256)
qkv_gemm_kernel(const bf16_t* __restrict__ xb,
                const bf16_t* __restrict__ wqb,
                const bf16_t* __restrict__ wkb,
                const bf16_t* __restrict__ wvb,
                bf16_t* __restrict__ qh,
                bf16_t* __restrict__ kh,
                bf16_t* __restrict__ vT) {
  __shared__ __align__(16) bf16_t As[2][128][64 + PAD];
  __shared__ __align__(16) bf16_t Bs[2][128][64 + PAD];

  const int tid  = threadIdx.x;
  const int lane = tid & 31;
  const int w    = tid >> 5;      // wave id 0..7
  const int n0   = lane & 15;
  const int half = lane >> 4;
  const int j0   = blockIdx.x << 7;   // 128 output columns
  const int m0   = blockIdx.y << 7;   // 128 output rows
  const int which = blockIdx.z;       // 0=q 1=k 2=v
  const bf16_t* __restrict__ W = (which == 0) ? wqb : (which == 1 ? wkb : wvb);

  v8f acc[8] = {};

  auto issue = [&](int buf, int kk) {
    // A tile + B tile: each 128 rows x 128B = 1024 chunks of 16B; 4 each/thread
#pragma unroll
    for (int i = 0; i < 4; ++i) {
      const int c   = tid + (i << 8);
      const int row = c >> 3, sub = c & 7;
      async16((unsigned)(uintptr_t)&As[buf][row][sub * 8],
              xb + (size_t)(m0 + row) * D_ + kk + sub * 8);
      async16((unsigned)(uintptr_t)&Bs[buf][row][sub * 8],
              W + (size_t)(j0 + row) * D_ + kk + sub * 8);
    }
  };

  issue(0, 0);
  for (int kk = 0; kk < D_; kk += 64) {
    const int buf = (kk >> 6) & 1;
    wait_async0();
    __syncthreads();
    if (kk + 64 < D_) issue(buf ^ 1, kk + 64);

#pragma unroll
    for (int kc = 0; kc < 64; kc += 32) {
      v16bf a = pack16(*(const v8bf*)&As[buf][w * 16 + n0][kc + half * 16],
                       *(const v8bf*)&As[buf][w * 16 + n0][kc + half * 16 + 8]);
      // software-pipelined B fragments: load nc+1 before consuming nc
      v16bf bcur = load16(&Bs[buf][n0][kc + half * 16]);
#pragma unroll
      for (int nc = 0; nc < 8; ++nc) {
        v16bf bnext = bcur;
        if (nc < 7) bnext = load16(&Bs[buf][(nc + 1) * 16 + n0][kc + half * 16]);
        acc[nc] = wmma_bf16(a, bcur, acc[nc]);
        bcur = bnext;
      }
    }
    __syncthreads();
  }

  // C tile layout: lane holds column n0; rows m = half*8 + r
  const int mrow0 = m0 + w * 16;
  const int bidx  = mrow0 >> 11;                       // batch
  const int sbase = (mrow0 & (S_ - 1)) + (half << 3);  // token index base

  if (which < 2) {  // ---- Q / K with fused RoPE ----
    bf16_t* dst = (which == 0) ? qh : kh;
#pragma unroll
    for (int nc = 0; nc < 8; ++nc) {
      const int   j  = j0 + nc * 16 + n0;
      const int   h  = j >> 7;
      const int   dk = j & (DK_ - 1);
      const float inv_freq = __expf(-(float)(dk >> 1) * 0.14391156831f); // ln(1e4)/64
      bf16_t* drow = dst + ((size_t)(bidx * H_ + h) * S_) * DK_ + dk;
#pragma unroll
      for (int r = 0; r < 8; ++r) {
        const int   s    = sbase + r;
        const float self = acc[nc][r];
        const float part = __shfl_xor(self, 1, 32);  // partner column j^1
        float sn, cs;
        __sincosf((float)s * inv_freq, &sn, &cs);
        const float res = (dk & 1) ? (part * sn + self * cs)
                                   : (self * cs - part * sn);
        drow[(size_t)s * DK_] = (bf16_t)res;
      }
    }
  } else {          // ---- V, stored transposed [bh][dk][s] ----
#pragma unroll
    for (int nc = 0; nc < 8; ++nc) {
      const int j  = j0 + nc * 16 + n0;
      const int h  = j >> 7;
      const int dk = j & (DK_ - 1);
      v8bf pk;
#pragma unroll
      for (int r = 0; r < 8; ++r) pk[r] = (bf16_t)acc[nc][r];
      *(v8bf*)(vT + ((size_t)(bidx * H_ + h) * DK_ + dk) * S_ + sbase) = pk;
    }
  }
}

// ---------------------------------------------------------------------------
// Flash attention, transposed-score formulation, 4 waves / block.
// Wave w handles q-tile blockIdx.x*4+w; the block shares double-buffered,
// async-staged K (32x128) and V^T (128x32) tiles in LDS. S^T = K*Q^T keeps
// softmax per-lane (+1 shfl_xor(16)); the S^T C-registers are bit-identical
// to the A-fragment of P for the P*V WMMA. Fragment loads are software-
// pipelined one ahead of the consuming WMMA.
// ---------------------------------------------------------------------------
__global__ void __launch_bounds__(128)
attn_kernel(const bf16_t* __restrict__ qh,
            const bf16_t* __restrict__ kh,
            const bf16_t* __restrict__ vT,
            float* __restrict__ out) {
  __shared__ __align__(16) bf16_t Kt[2][32][DK_ + PAD];
  __shared__ __align__(16) bf16_t Vt[2][DK_][32 + PAD];

  const int tid  = threadIdx.x;
  const int lane = tid & 31;
  const int w    = tid >> 5;
  const int n0   = lane & 15;
  const int half = lane >> 4;
  const int bh   = blockIdx.y;
  const int bidx = bh >> 4;
  const int h    = bh & 15;
  const int qit  = blockIdx.x * 4 + w;        // this wave's q tile (0..127)
  const size_t head  = (size_t)bh * S_ * DK_;
  const bf16_t* __restrict__ vhead = vT + (size_t)bh * DK_ * S_;

  // Persistent Q B-fragments: B[k][n] = q[qrow=n][d=k]
  v16bf qB[4];
  {
    const bf16_t* qrow = qh + head + (size_t)(qit * 16 + n0) * DK_;
#pragma unroll
    for (int c = 0; c < 4; ++c) qB[c] = load16(qrow + c * 32 + (half << 4));
  }

  v8f   O[8] = {};
  float mrun = -1e30f, lrun = 0.f;
  const float scale = 0.08838834764831845f;   // 1/sqrt(128)
  const int   qrow  = qit * 16 + n0;
  const int   qend_blk = (blockIdx.x + 1) * 64;   // uniform loop bound for block

  auto issue = [&](int buf, int j32) {
    // K tile: 32 rows x 256B = 512 chunks; V tile: 128 rows x 64B = 512 chunks.
#pragma unroll
    for (int i = 0; i < 4; ++i) {
      const int c = tid + (i << 7);
      const int krow = c >> 4, ksub = c & 15;
      async16((unsigned)(uintptr_t)&Kt[buf][krow][ksub * 8],
              kh + head + (size_t)(j32 + krow) * DK_ + ksub * 8);
      const int vrow = c >> 2, vsub = c & 3;
      async16((unsigned)(uintptr_t)&Vt[buf][vrow][vsub * 8],
              vhead + (size_t)vrow * S_ + j32 + vsub * 8);
    }
  };

  // K A-fragment loader: f = tile*4 + c  (tile 0: keys 0..15, tile 1: 16..31)
  auto loadK = [&](int buf, int f) -> v16bf {
    const int row = ((f >> 2) << 4) + n0;
    const int c   = f & 3;
    return pack16(*(const v8bf*)&Kt[buf][row][c * 32 + half * 16],
                  *(const v8bf*)&Kt[buf][row][c * 32 + half * 16 + 8]);
  };

  issue(0, 0);
  for (int j32 = 0; j32 < qend_blk; j32 += 32) {
    const int buf = (j32 >> 5) & 1;
    wait_async0();
    __syncthreads();
    if (j32 + 32 < qend_blk) issue(buf ^ 1, j32 + 32);

    // --- scores^T with pipelined K fragments ---
    v8f T0 = {}, T1 = {};
    v16bf kcur = loadK(buf, 0);
#pragma unroll
    for (int f = 0; f < 8; ++f) {
      v16bf knext = kcur;
      if (f < 7) knext = loadK(buf, f + 1);
      if (f < 4) T0 = wmma_bf16(kcur, qB[f], T0);
      else       T1 = wmma_bf16(kcur, qB[f - 4], T1);
      kcur = knext;
    }

    // --- scale + causal mask (mask only at/past this wave's diagonal) ---
    float s0[8], s1[8];
    const bool domask = (j32 + 31 > qit * 16);
#pragma unroll
    for (int r = 0; r < 8; ++r) {
      float v0 = T0[r] * scale;
      float v1 = T1[r] * scale;
      if (domask) {
        const int key = j32 + (half << 3) + r;
        if (key > qrow)      v0 = -1e9f;
        if (key + 16 > qrow) v1 = -1e9f;
      }
      s0[r] = v0; s1[r] = v1;
    }

    // --- online softmax (lane + lane^16 jointly own query row n0) ---
    float mloc = s0[0];
#pragma unroll
    for (int r = 0; r < 8; ++r) { mloc = fmaxf(mloc, s0[r]); mloc = fmaxf(mloc, s1[r]); }
    mloc = fmaxf(mloc, __shfl_xor(mloc, 16, 32));
    const float mnew  = fmaxf(mrun, mloc);
    const float alpha = __expf(mrun - mnew);

    v16bf pA;   // P already in A-fragment layout by construction
    float psum = 0.f;
#pragma unroll
    for (int r = 0; r < 8; ++r) {
      const float p0 = __expf(s0[r] - mnew);
      const float p1 = __expf(s1[r] - mnew);
      psum += p0 + p1;
      pA[r]     = (bf16_t)p0;
      pA[r + 8] = (bf16_t)p1;
    }
    psum += __shfl_xor(psum, 16, 32);
    lrun = lrun * alpha + psum;
    mrun = mnew;

    float alphas[8];
#pragma unroll
    for (int r = 0; r < 8; ++r) alphas[r] = __shfl(alpha, (half << 3) + r, 32);
#pragma unroll
    for (int nc = 0; nc < 8; ++nc)
#pragma unroll
      for (int r = 0; r < 8; ++r) O[nc][r] *= alphas[r];

    // --- O += P * V from the staged V^T tile (pipelined B fragments) ---
    v16bf vcur = pack16(*(const v8bf*)&Vt[buf][n0][half << 3], *(const v8bf*)&Vt[buf][n0][16 + (half << 3)]);
#pragma unroll
    for (int nc = 0; nc < 8; ++nc) {
      v16bf vnext = vcur;
      if (nc < 7) vnext = pack16(*(const v8bf*)&Vt[buf][(nc + 1) * 16 + n0][half << 3], *(const v8bf*)&Vt[buf][(nc + 1) * 16 + n0][16 + (half << 3)]);
      O[nc] = wmma_bf16(pA, vcur, O[nc]);
      vcur = vnext;
    }
    __syncthreads();
  }

  // --- normalize and store: out[b][s][h*128 + d], f32 ---
  float linv[8];
#pragma unroll
  for (int r = 0; r < 8; ++r) linv[r] = 1.f / __shfl(lrun, (half << 3) + r, 32);
#pragma unroll
  for (int nc = 0; nc < 8; ++nc) {
#pragma unroll
    for (int r = 0; r < 8; ++r) {
      const int s = qit * 16 + (half << 3) + r;
      out[((size_t)(bidx * S_ + s)) * D_ + h * DK_ + nc * 16 + n0] = O[nc][r] * linv[r];
    }
  }
}

// ---------------------------------------------------------------------------
extern "C" void kernel_launch(void* const* d_in, const int* in_sizes, int n_in,
                              void* d_out, int out_size, void* d_ws, size_t ws_size,
                              hipStream_t stream) {
  const float* x  = (const float*)d_in[0];
  const float* wq = (const float*)d_in[1];
  const float* wk = (const float*)d_in[2];
  const float* wv = (const float*)d_in[3];
  float* out = (float*)d_out;

  const size_t nx = (size_t)B_ * S_ * D_;   // 8,388,608
  const size_t nw = (size_t)D_ * D_;        // 4,194,304

  bf16_t* xb  = (bf16_t*)d_ws;
  bf16_t* wqb = xb  + nx;
  bf16_t* wkb = wqb + nw;
  bf16_t* wvb = wkb + nw;
  bf16_t* qh  = wvb + nw;   // [B*H][S][DK]
  bf16_t* kh  = qh  + nx;   // [B*H][S][DK]
  bf16_t* vT  = kh  + nx;   // [B*H][DK][S]

  cvt_f32_bf16<<<dim3((unsigned)((nx + 255) / 256)), 256, 0, stream>>>(x,  xb,  (int)nx);
  cvt_f32_bf16<<<dim3((unsigned)((nw + 255) / 256)), 256, 0, stream>>>(wq, wqb, (int)nw);
  cvt_f32_bf16<<<dim3((unsigned)((nw + 255) / 256)), 256, 0, stream>>>(wk, wkb, (int)nw);
  cvt_f32_bf16<<<dim3((unsigned)((nw + 255) / 256)), 256, 0, stream>>>(wv, wvb, (int)nw);

  qkv_gemm_kernel<<<dim3(D_ / 128, (B_ * S_) / 128, 3), 256, 0, stream>>>(
      xb, wqb, wkb, wvb, qh, kh, vT);

  attn_kernel<<<dim3(S_ / 64, B_ * H_), 128, 0, stream>>>(qh, kh, vT, out);
}
